// CondScoreModelGNN_67327907332484
// MI455X (gfx1250) — matrix operand order repaired and checked by reading, not compiled, not run on hardware
//
#include <hip/hip_runtime.h>
#include <hip/hip_bf16.h>
#include <math.h>

// ---------------- CDNA5 WMMA types ----------------
typedef __attribute__((ext_vector_type(16))) _Float16 v16h;
typedef __attribute__((ext_vector_type(8)))  _Float16 v8h;
typedef __attribute__((ext_vector_type(8)))  float    v8f;

union AU { v16h v; v8h h8[2]; };

__device__ __forceinline__ v8f wmma16(v16h a, v16h b, v8f c) {
  // D = A(16x32 f16) x B(32x16 f16) + C(16x16 f32)
  return __builtin_amdgcn_wmma_f32_16x16x32_f16(
      /*neg_a=*/false, a, /*neg_b=*/false, b,
      /*c_mod=*/(short)0, c, /*reuse_a=*/false, /*reuse_b=*/false);
}

// generic-pointer low 32 bits = LDS byte offset (ISA 10.2 aperture rules)
__device__ __forceinline__ unsigned lds_off(const void* p) {
  return (unsigned)(unsigned long long)(uintptr_t)p;
}

// ---- monotonic float<->uint key for atomic segment-max ----
__device__ __forceinline__ unsigned fkey(float f) {
  unsigned u = __float_as_uint(f);
  return (u & 0x80000000u) ? ~u : (u | 0x80000000u);
}
__device__ __forceinline__ float unfkey(unsigned k) {
  unsigned u = (k & 0x80000000u) ? (k ^ 0x80000000u) : ~k;
  return __uint_as_float(u);
}

#define TWO_PI_F 6.283185307179586f

// =====================================================================
// 0) fill uint buffer (init segment-max keys; key 0 < key(-inf))
// =====================================================================
__global__ void k_fill_u32(unsigned* __restrict__ p, unsigned v, long long n) {
  long long i = (long long)blockIdx.x * blockDim.x + threadIdx.x;
  long long stride = (long long)gridDim.x * blockDim.x;
  for (; i < n; i += stride) p[i] = v;
}

// =====================================================================
// 1) wall MLP: (B,2)@ (2,64) relu @ (64,64) relu  -> wall_h (B,64)
// =====================================================================
__global__ void k_wall(const float* __restrict__ wall,
                       const float* __restrict__ w1, const float* __restrict__ wb1,
                       const float* __restrict__ w2, const float* __restrict__ wb2,
                       float* __restrict__ wall_h) {
  __shared__ float h1[64];
  int b = blockIdx.x, j = threadIdx.x;
  float v = wall[b * 2 + 0] * w1[0 * 64 + j] + wall[b * 2 + 1] * w1[1 * 64 + j] + wb1[j];
  h1[j] = v > 0.f ? v : 0.f;
  __syncthreads();
  float s = wb2[j];
#pragma unroll 8
  for (int k = 0; k < 64; ++k) s += h1[k] * w2[k * 64 + j];
  wall_h[b * 64 + j] = s > 0.f ? s : 0.f;
}

// =====================================================================
// 2) pack weights (K x Ncols, row-major f32) into WMMA B-operand panels.
// contiguous k order used here (same as A loads): N=lane&15, K = i + 16*(lane>>4) per
// 32-K tile. Panel layout: out[(((kt*ntiles)+nt)*32 + lane)*16 + i]
// =====================================================================
__global__ void k_pack_b(const float* __restrict__ W, const float* __restrict__ Wsub,
                         _Float16* __restrict__ out, int K, int Ncols) {
  int ktiles = K >> 5, ntiles = Ncols >> 4;
  int total = ktiles * ntiles * 32 * 16;
  for (int idx = blockIdx.x * blockDim.x + threadIdx.x; idx < total;
       idx += gridDim.x * blockDim.x) {
    int i    = idx & 15;
    int lane = (idx >> 4) & 31;
    int rest = idx >> 9;
    int nt = rest % ntiles;
    int kt = rest / ntiles;
    int k = kt * 32 + i + 16 * (lane >> 4);
    int n = nt * 16 + (lane & 15);
    float v = W[k * Ncols + n];
    if (Wsub) v -= Wsub[k * Ncols + n];
    out[idx] = (_Float16)v;
  }
}

// =====================================================================
// 3) per-node features -> X row (320 f16): [init(128)|class(64)|sigma(64)|wall(64)]
//    also inv_std[n] = 1/(marginal_prob_std(t)+1e-7)
// =====================================================================
__global__ void k_node_feat(const float* __restrict__ t,
                            const float* __restrict__ obj_x, const float* __restrict__ obj_geo,
                            const int* __restrict__ category, const int* __restrict__ batch_idx,
                            const float* __restrict__ embed_W, const float* __restrict__ gfp_W,
                            const float* __restrict__ sW, const float* __restrict__ sb,
                            const float* __restrict__ i1, const float* __restrict__ ib1,
                            const float* __restrict__ i2, const float* __restrict__ ib2,
                            const float* __restrict__ wall_h,
                            _Float16* __restrict__ X, float* __restrict__ inv_std) {
  __shared__ float gfp[64];
  __shared__ float h1[128];
  int n = blockIdx.x, tid = threadIdx.x;
  float tv = t[n];

  // Gaussian Fourier projection: [sin(t*w*2pi), cos(t*w*2pi)], w has 32 entries
  float xp = tv * gfp_W[tid & 31] * TWO_PI_F;
  gfp[tid] = (tid < 32) ? sinf(xp) : cosf(xp);

  // init MLP layer 1: xin(6) @ i1(6x128) + ib1, relu
  float xin[6];
#pragma unroll
  for (int k = 0; k < 4; ++k) xin[k] = obj_x[(long long)n * 4 + k];
#pragma unroll
  for (int k = 0; k < 2; ++k) xin[4 + k] = obj_geo[(long long)n * 2 + k];
#pragma unroll
  for (int jj = 0; jj < 2; ++jj) {
    int j = tid + jj * 64;
    float s = ib1[j];
#pragma unroll
    for (int k = 0; k < 6; ++k) s += xin[k] * i1[k * 128 + j];
    h1[j] = s > 0.f ? s : 0.f;
  }
  __syncthreads();

  _Float16* xr = X + (long long)n * 320;

  // sigma_feat = relu(gfp @ sW + sb)
  float sg = sb[tid];
#pragma unroll 8
  for (int k = 0; k < 64; ++k) sg += gfp[k] * sW[k * 64 + tid];
  sg = sg > 0.f ? sg : 0.f;

  // init MLP layer 2: h1(128) @ i2(128x128) + ib2, relu
#pragma unroll
  for (int jj = 0; jj < 2; ++jj) {
    int j = tid + jj * 64;
    float s2 = ib2[j];
#pragma unroll 8
    for (int k = 0; k < 128; ++k) s2 += h1[k] * i2[k * 128 + j];
    xr[j] = (_Float16)(s2 > 0.f ? s2 : 0.f);
  }

  // class / sigma / wall segments of cond
  int cat = category[n];
  int bi  = batch_idx[n];
  float cf = embed_W[cat * 64 + tid];
  xr[128 + tid] = (_Float16)(cf > 0.f ? cf : 0.f);
  xr[192 + tid] = (_Float16)sg;
  xr[256 + tid] = (_Float16)wall_h[bi * 64 + tid];

  if (tid == 0) {
    const float ln_sigma = 3.2188758248682006f; // ln(25)
    float stdv = sqrtf((expf(2.f * tv * ln_sigma) - 1.f) / (2.f * ln_sigma));
    inv_std[n] = 1.f / (stdv + 1e-7f);
  }
}

// =====================================================================
// 4) node GEMM: hc = X @ Wd (320x128), hn = X @ Wb.
//    64-row macro tile; A staged in LDS via async-to-LDS. 8 waves:
//    wave[2]=matrix (D->hc / B->hn), wave[1:0]=N-tile pair.
//    acc[4][2] (64 VGPRs) + 2 B tiles (16) + A (8): spill-free, with B
//    amortized over 4 row-tiles.
// =====================================================================
__global__ void k_node_gemm(const _Float16* __restrict__ X,
                            const _Float16* __restrict__ packD,
                            const _Float16* __restrict__ packB,
                            _Float16* __restrict__ hc, _Float16* __restrict__ hn,
                            int N) {
  __shared__ _Float16 xs[64 * 320];       // 40 KB A panel
  int tid = threadIdx.x;                  // 256 threads = 8 waves
  int row0 = blockIdx.x * 64;

  // async-stage 64 rows of X (40960 B): 8 waves x 10 iters x 32 lanes x 16 B
  {
    const _Float16* gsrc = X + (long long)row0 * 320;
    unsigned lbase = lds_off(&xs[0]);
#pragma unroll
    for (int it = 0; it < 10; ++it) {
      unsigned off = (unsigned)(it * 256 + tid) * 16u;
      asm volatile("global_load_async_to_lds_b128 %0, %1, %2"
                   :: "v"(lbase + off), "v"(off), "s"(gsrc) : "memory");
    }
  }
  asm volatile("s_wait_asynccnt 0x0" ::: "memory");
  __syncthreads();

  int wave = tid >> 5, lane = tid & 31;
  const _Float16* pack = (wave & 4) ? packB : packD;
  _Float16* out = (wave & 4) ? hn : hc;
  int nbase = (wave & 3) * 2;             // this wave's 2 N-tiles
  int m = lane & 15, h = lane >> 4;

  v8f acc[4][2] = {};                     // [row-tile][n-tile]
#pragma unroll
  for (int kt = 0; kt < 10; ++kt) {
    v16h b[2];
#pragma unroll
    for (int i = 0; i < 2; ++i)
      b[i] = *(const v16h*)(pack + ((((kt * 8) + nbase + i) * 32 + lane) << 4));
    if (kt < 9)
      __builtin_prefetch(pack + ((((kt + 1) * 8 + nbase) * 32 + lane) << 4), 0, 1);
#pragma unroll
    for (int rt = 0; rt < 4; ++rt) {
      // 16-bit A layout: lane holds M=lane&15;
      // K = {16h..16h+15} within each 32-wide K tile (matches packed B).
      AU a;
      const _Float16* ap = xs + (rt * 16 + m) * 320 + kt * 32 + 16 * h;
      a.h8[0] = *(const v8h*)ap;
      a.h8[1] = *(const v8h*)(ap + 8);
#pragma unroll
      for (int i = 0; i < 2; ++i)
        acc[rt][i] = wmma16(a.v, b[i], acc[rt][i]);
    }
  }
  // D layout: lane<16 -> N=lane, M=r ; lane>=16 -> N=lane-16, M=r+8
#pragma unroll
  for (int rt = 0; rt < 4; ++rt)
#pragma unroll
    for (int i = 0; i < 2; ++i)
#pragma unroll
      for (int r = 0; r < 8; ++r) {
        int rr = row0 + rt * 16 + r + 8 * h;
        if (rr < N)
          out[(long long)rr * 128 + (nbase + i) * 16 + (lane & 15)] =
              (_Float16)acc[rt][i][r];
      }
}

// =====================================================================
// 5) edge conv1: 64 edges/block, 8 waves. W2 panel async-staged to LDS
//    overlapped with the gather/ReLU phase. Phase 2: wave w handles
//    (edge-tile w>>1, N-tile half w&1): acc[4] = 32 VGPRs, WMMA from LDS,
//    epilogue +b2 then atomic segment-max.
// =====================================================================
__global__ void k_edge_conv1(const _Float16* __restrict__ hc, const _Float16* __restrict__ hn,
                             const int* __restrict__ src, const int* __restrict__ dst,
                             const float* __restrict__ b1,
                             const _Float16* __restrict__ packW2, const float* __restrict__ b2,
                             unsigned* __restrict__ ykeys, int E) {
  __shared__ _Float16 et[64 * 128];       // 16 KB edge rows (f16)
  __shared__ _Float16 bsm[128 * 128];     // 32 KB packed W2 panels
  __shared__ int sdst[64];
  int tid = threadIdx.x;                  // 256 threads = 8 waves
  int eb = blockIdx.x * 64;

  // kick off async copy of W2 panels (32768 B = 256 thr * 8 iters * 16 B)
  {
    unsigned lbase = lds_off(&bsm[0]);
#pragma unroll
    for (int it = 0; it < 8; ++it) {
      unsigned off = (unsigned)(it * 256 + tid) * 16u;
      asm volatile("global_load_async_to_lds_b128 %0, %1, %2"
                   :: "v"(lbase + off), "v"(off), "s"(packW2) : "memory");
    }
  }

  // phase 1 (overlaps async copy): 4 threads per edge, 32 cols each
  int le = tid >> 2;
  int c0 = (tid & 3) * 32;
  int e = eb + le; if (e >= E) e = E - 1;  // dup edge -> idempotent under max
  int s = src[e], d = dst[e];
  if ((tid & 3) == 0) sdst[le] = d;
  const v8h* hcp = (const v8h*)(hc + (long long)d * 128 + c0);
  const v8h* hnp = (const v8h*)(hn + (long long)s * 128 + c0);
#pragma unroll
  for (int q = 0; q < 4; ++q) {
    v8h a = hcp[q], bq = hnp[q];
#pragma unroll
    for (int j = 0; j < 8; ++j) {
      int c = c0 + q * 8 + j;
      float v = (float)a[j] + (float)bq[j] + b1[c];
      et[le * 128 + c] = (_Float16)(v > 0.f ? v : 0.f);
    }
  }
  asm volatile("s_wait_asynccnt 0x0" ::: "memory");
  __syncthreads();

  // phase 2
  int wave = tid >> 5, lane = tid & 31;
  int etile = wave >> 1;                  // 0..3 (16 edges each)
  int nbase = (wave & 1) * 4;             // this wave's 4 N-tiles
  int m = lane & 15, h = lane >> 4;
  const _Float16* etw = et + etile * 16 * 128;
  v8f acc[4] = {};
#pragma unroll
  for (int kt = 0; kt < 4; ++kt) {
    AU a;
    const _Float16* ap = etw + m * 128 + kt * 32 + 16 * h;
    a.h8[0] = *(const v8h*)ap;
    a.h8[1] = *(const v8h*)(ap + 8);
#pragma unroll
    for (int i = 0; i < 4; ++i) {
      int nt = nbase + i;
      v16h b = *(const v16h*)(bsm + ((((kt * 8) + nt) * 32 + lane) << 4));
      acc[i] = wmma16(a.v, b, acc[i]);
    }
  }
#pragma unroll
  for (int i = 0; i < 4; ++i) {
    int nt = nbase + i;
#pragma unroll
    for (int r = 0; r < 8; ++r) {
      int mm = r + 8 * h;
      int col = nt * 16 + (lane & 15);
      float v = acc[i][r] + b2[col];
      atomicMax(ykeys + (long long)sdst[etile * 16 + mm] * 128 + col, fkey(v));
    }
  }
}

// =====================================================================
// 6) writeback: X[:, :128] = f16(relu(unkey(y1)))
// =====================================================================
__global__ void k_relu_wb(const unsigned* __restrict__ y1, _Float16* __restrict__ X,
                          long long n128) {
  long long i = (long long)blockIdx.x * blockDim.x + threadIdx.x;
  long long stride = (long long)gridDim.x * blockDim.x;
  for (; i < n128; i += stride) {
    float f = unfkey(y1[i]);
    f = f > 0.f ? f : 0.f;
    X[(i >> 7) * 320 + (i & 127)] = (_Float16)f;
  }
}

// =====================================================================
// 7) edge conv2: e = relu(hc[dst]+hn[src]+b1), e @ W2(128x4)+b2, seg-max.
//    Tiny N dim -> VALU dot products from LDS.
// =====================================================================
__global__ void k_edge_conv2(const _Float16* __restrict__ hc, const _Float16* __restrict__ hn,
                             const int* __restrict__ src, const int* __restrict__ dst,
                             const float* __restrict__ b1,
                             const float* __restrict__ W2, const float* __restrict__ b2,
                             unsigned* __restrict__ ykeys, int E) {
  __shared__ _Float16 et[16 * 128];
  __shared__ int sdst[16];
  int tid = threadIdx.x;
  int eb = blockIdx.x * 16;

  int le = tid >> 2;
  int c0 = (tid & 3) * 32;
  int e = eb + le; if (e >= E) e = E - 1;
  int s = src[e], d = dst[e];
  if ((tid & 3) == 0) sdst[le] = d;
  const v8h* hcp = (const v8h*)(hc + (long long)d * 128 + c0);
  const v8h* hnp = (const v8h*)(hn + (long long)s * 128 + c0);
#pragma unroll
  for (int q = 0; q < 4; ++q) {
    v8h a = hcp[q], b = hnp[q];
#pragma unroll
    for (int j = 0; j < 8; ++j) {
      int c = c0 + q * 8 + j;
      float v = (float)a[j] + (float)b[j] + b1[c];
      et[le * 128 + c] = (_Float16)(v > 0.f ? v : 0.f);
    }
  }
  __syncthreads();

  int le2 = tid >> 2, o = tid & 3;
  float sum = b2[o];
#pragma unroll 8
  for (int k = 0; k < 128; ++k) sum += (float)et[le2 * 128 + k] * W2[k * 4 + o];
  atomicMax(ykeys + (long long)sdst[le2] * 4 + o, fkey(sum));
}

// =====================================================================
// 8) finalize: out = unkey(y2) * inv_std[n]
// =====================================================================
__global__ void k_final(const unsigned* __restrict__ y2, const float* __restrict__ inv_std,
                        float* __restrict__ out, long long n4) {
  long long i = (long long)blockIdx.x * blockDim.x + threadIdx.x;
  long long stride = (long long)gridDim.x * blockDim.x;
  for (; i < n4; i += stride) out[i] = unfkey(y2[i]) * inv_std[i >> 2];
}

// =====================================================================
extern "C" void kernel_launch(void* const* d_in, const int* in_sizes, int n_in,
                              void* d_out, int out_size, void* d_ws, size_t ws_size,
                              hipStream_t stream) {
  (void)n_in; (void)out_size; (void)ws_size;
  const float* t_in     = (const float*)d_in[0];
  const float* obj_x    = (const float*)d_in[1];
  const float* obj_geo  = (const float*)d_in[2];
  const float* wall     = (const float*)d_in[3];
  const int*   category = (const int*)d_in[4];
  const int*   batch_idx= (const int*)d_in[5];
  const int*   src      = (const int*)d_in[6];
  const int*   dst      = (const int*)d_in[7];
  const float* embed_W  = (const float*)d_in[8];
  const float* gfp_W    = (const float*)d_in[9];
  const float* sW       = (const float*)d_in[10];
  const float* sb       = (const float*)d_in[11];
  const float* w1       = (const float*)d_in[12];
  const float* wb1      = (const float*)d_in[13];
  const float* w2       = (const float*)d_in[14];
  const float* wb2      = (const float*)d_in[15];
  const float* i1       = (const float*)d_in[16];
  const float* ib1      = (const float*)d_in[17];
  const float* i2       = (const float*)d_in[18];
  const float* ib2      = (const float*)d_in[19];
  const float* m1W1     = (const float*)d_in[20];
  const float* m1b1     = (const float*)d_in[21];
  const float* m1W2     = (const float*)d_in[22];
  const float* m1b2     = (const float*)d_in[23];
  const float* m2W1     = (const float*)d_in[24];
  const float* m2b1     = (const float*)d_in[25];
  const float* m2W2     = (const float*)d_in[26];
  const float* m2b2     = (const float*)d_in[27];
  float* out = (float*)d_out;

  const int N = in_sizes[0];          // t is (N,1)
  const int E = in_sizes[6];          // src
  const int B = in_sizes[3] / 2;      // wall is (B,2)
  const int Npad = (N + 63) & ~63;    // X padded so 64-row async staging stays in ws

  // ---- carve workspace ----
  unsigned char* p = (unsigned char*)d_ws;
  auto carve = [&](size_t bytes) -> void* {
    void* r = (void*)p;
    p += (bytes + 255) & ~(size_t)255;
    return r;
  };
  _Float16* X    = (_Float16*)carve((size_t)Npad * 320 * 2);
  _Float16* hcB  = (_Float16*)carve((size_t)N * 128 * 2);
  _Float16* hnB  = (_Float16*)carve((size_t)N * 128 * 2);
  unsigned* y1   = (unsigned*)carve((size_t)N * 128 * 4);
  unsigned* y2   = (unsigned*)carve((size_t)N * 4 * 4);
  float*    wallh= (float*)carve((size_t)B * 64 * 4);
  float*    invs = (float*)carve((size_t)N * 4);
  _Float16* p1d  = (_Float16*)carve((size_t)320 * 128 * 2);
  _Float16* p1b  = (_Float16*)carve((size_t)320 * 128 * 2);
  _Float16* pw2  = (_Float16*)carve((size_t)128 * 128 * 2);
  _Float16* p2d  = (_Float16*)carve((size_t)320 * 128 * 2);
  _Float16* p2b  = (_Float16*)carve((size_t)320 * 128 * 2);

  // ---- init segment-max key buffers ----
  {
    long long n1 = (long long)N * 128, n2 = (long long)N * 4;
    k_fill_u32<<<dim3((unsigned)((n1 + 255) / 256)), 256, 0, stream>>>(y1, 0u, n1);
    k_fill_u32<<<dim3((unsigned)((n2 + 255) / 256)), 256, 0, stream>>>(y2, 0u, n2);
  }

  // ---- pack weights into WMMA B panels ----
  k_pack_b<<<dim3(160), 256, 0, stream>>>(m1W1,               m1W1 + 320 * 128, p1d, 320, 128);
  k_pack_b<<<dim3(160), 256, 0, stream>>>(m1W1 + 320 * 128,   nullptr,          p1b, 320, 128);
  k_pack_b<<<dim3(64),  256, 0, stream>>>(m1W2,               nullptr,          pw2, 128, 128);
  k_pack_b<<<dim3(160), 256, 0, stream>>>(m2W1,               m2W1 + 320 * 128, p2d, 320, 128);
  k_pack_b<<<dim3(160), 256, 0, stream>>>(m2W1 + 320 * 128,   nullptr,          p2b, 320, 128);

  // ---- wall MLP ----
  k_wall<<<dim3(B), 64, 0, stream>>>(wall, w1, wb1, w2, wb2, wallh);

  // ---- per-node features -> X (N x 320 f16), inv_std ----
  k_node_feat<<<dim3(N), 64, 0, stream>>>(t_in, obj_x, obj_geo, category, batch_idx,
                                          embed_W, gfp_W, sW, sb, i1, ib1, i2, ib2,
                                          wallh, X, invs);

  const int ntile64 = (N + 63) / 64;
  const int etile64 = (E + 63) / 64;
  const int etile16 = (E + 15) / 16;

  // ---- conv1: node GEMMs, edge GEMM + seg-max ----
  k_node_gemm<<<dim3(ntile64), 256, 0, stream>>>(X, p1d, p1b, hcB, hnB, N);
  k_edge_conv1<<<dim3(etile64), 256, 0, stream>>>(hcB, hnB, src, dst, m1b1, pw2, m1b2, y1, E);

  // ---- relu + writeback into X[:, :128] (cond part of X is reused) ----
  {
    long long n1 = (long long)N * 128;
    k_relu_wb<<<dim3((unsigned)((n1 + 255) / 256)), 256, 0, stream>>>(y1, X, n1);
  }

  // ---- conv2: node GEMMs, edge dot + seg-max ----
  k_node_gemm<<<dim3(ntile64), 256, 0, stream>>>(X, p2d, p2b, hcB, hnB, N);
  k_edge_conv2<<<dim3(etile16), 64, 0, stream>>>(hcB, hnB, src, dst, m2b1, m2W2, m2b2, y2, E);

  // ---- finalize ----
  {
    long long n4 = (long long)N * 4;
    k_final<<<dim3((unsigned)((n4 + 255) / 256)), 256, 0, stream>>>(y2, invs, out, n4);
  }
}
